// BlockPheno_32564442038622
// MI455X (gfx1250) — matrix operand order repaired and checked by reading, not compiled, not run on hardware
//
#include <hip/hip_runtime.h>

typedef __bf16 bf16;
typedef __attribute__((ext_vector_type(16))) __bf16 v16bf;
typedef __attribute__((ext_vector_type(8)))  float  v8f;
typedef __attribute__((ext_vector_type(4)))  int    v4i;

union Frag { v16bf v; uint4 u[2]; };

// ---- CDNA5 async global->LDS path (guarded: falls back to reg-staged copies)
#if defined(__AMDGCN__) && __has_builtin(__builtin_amdgcn_global_load_async_to_lds_b128)
#define ASYNC_LDS 1
#else
#define ASYNC_LDS 0
#endif

#if ASYNC_LDS
typedef __attribute__((address_space(1))) v4i gv4i;  // global 16B vector
typedef __attribute__((address_space(3))) v4i lv4i;  // LDS 16B vector
__device__ __forceinline__ void async_copy16(const bf16* g, bf16* l) {
  __builtin_amdgcn_global_load_async_to_lds_b128((gv4i*)g, (lv4i*)l, 0, 0);
}
__device__ __forceinline__ void async_wait0() {
#if __has_builtin(__builtin_amdgcn_s_wait_asynccnt)
  __builtin_amdgcn_s_wait_asynccnt(0);
#else
  asm volatile("s_wait_asynccnt 0x0" ::: "memory");
#endif
}
#endif

__device__ __forceinline__ bf16 f2bf(float f) {
  union { float f; unsigned u; } a; a.f = f;
  unsigned r = a.u + 0x7FFFu + ((a.u >> 16) & 1u);  // round-to-nearest-even
  unsigned short h = (unsigned short)(r >> 16);
  union { unsigned short s; bf16 b; } o; o.s = h;
  return o.b;
}

// ---------------------------------------------------------------------------
// elementwise fp32 -> bf16 cast (weights)
// ---------------------------------------------------------------------------
__global__ __launch_bounds__(256) void cast_f32_to_bf16(
    const float* __restrict__ in, bf16* __restrict__ out, int n) {
  int i = blockIdx.x * 256 + threadIdx.x;
  if (i < n) out[i] = f2bf(in[i]);
}

// ---------------------------------------------------------------------------
// LayerNorm over D=1024, one block (256 thr) per row, bf16 output
// ---------------------------------------------------------------------------
__global__ __launch_bounds__(256) void layernorm_to_bf16(
    const float* __restrict__ x, const float* __restrict__ g,
    const float* __restrict__ bta, bf16* __restrict__ out) {
  __shared__ float red[16];
  const int row = blockIdx.x, t = threadIdx.x;
  const float* xr = x + (size_t)row * 1024;
  float v[4], s = 0.f, sq = 0.f;
#pragma unroll
  for (int i = 0; i < 4; ++i) {
    v[i] = xr[t + 256 * i];
    s += v[i]; sq += v[i] * v[i];
  }
#pragma unroll
  for (int off = 16; off > 0; off >>= 1) {
    s  += __shfl_xor(s,  off, 32);
    sq += __shfl_xor(sq, off, 32);
  }
  const int wid = t >> 5;
  if ((t & 31) == 0) { red[wid] = s; red[8 + wid] = sq; }
  __syncthreads();
  s = 0.f; sq = 0.f;
#pragma unroll
  for (int w = 0; w < 8; ++w) { s += red[w]; sq += red[8 + w]; }
  const float mean = s * (1.0f / 1024.0f);
  const float var  = sq * (1.0f / 1024.0f) - mean * mean;
  const float rstd = rsqrtf(var + 1e-5f);
#pragma unroll
  for (int i = 0; i < 4; ++i) {
    int c = t + 256 * i;
    out[(size_t)row * 1024 + c] = f2bf((v[i] - mean) * rstd * g[c] + bta[c]);
  }
}

// ---------------------------------------------------------------------------
// C[M,N] = A[M,K] * W[N,K]^T  (bf16 in, f32 accum), fused epilogue.
// Tile 128x128x32, double-buffered LDS, software-pipelined loads
// (async global->LDS when available, else reg-staged). 8 wave32 waves (4x2),
// each wave: 2x4 tiles of v_wmma_f32_16x16x32_bf16.
// ---------------------------------------------------------------------------
template <bool RELU, bool HAS_RES, bool OUT_BF16>
__global__ __launch_bounds__(256) void gemm_bf16_wmma(
    const bf16* __restrict__ A, const bf16* __restrict__ W,
    const float* __restrict__ bias, const float* __restrict__ res,
    void* __restrict__ outv, int M, int N, int K) {
  __shared__ bf16 As[2][128][40];   // 80B row stride: 16B-aligned, conflict-free
  __shared__ bf16 Bs[2][128][40];
  const int t = threadIdx.x;
  const int wid = t >> 5, lane = t & 31;
  const int wm = wid & 3, wn = wid >> 2;
  const int lhalf = lane >> 4, l16 = lane & 15;
  const int m_blk = blockIdx.y * 128, n_blk = blockIdx.x * 128;
  const int KT = K >> 5;

#if ASYNC_LDS
  auto issue_tile = [&](int k0, int buf) {
#pragma unroll
    for (int i = 0; i < 2; ++i) {
      int id = t + 256 * i;                  // 512 16B chunks per matrix
      int row = id >> 2, ch = (id & 3) * 8;
      async_copy16(A + (size_t)(m_blk + row) * K + k0 + ch, &As[buf][row][ch]);
      async_copy16(W + (size_t)(n_blk + row) * K + k0 + ch, &Bs[buf][row][ch]);
    }
  };
  issue_tile(0, 0);
  async_wait0();
  __syncthreads();
#else
  uint4 ra[2], rb[2];
  auto fetch_tile = [&](int k0) {
#pragma unroll
    for (int i = 0; i < 2; ++i) {
      int id = t + 256 * i;
      int row = id >> 2, ch = (id & 3) * 8;
      ra[i] = *reinterpret_cast<const uint4*>(A + (size_t)(m_blk + row) * K + k0 + ch);
      rb[i] = *reinterpret_cast<const uint4*>(W + (size_t)(n_blk + row) * K + k0 + ch);
    }
  };
  auto commit_tile = [&](int buf) {
#pragma unroll
    for (int i = 0; i < 2; ++i) {
      int id = t + 256 * i;
      int row = id >> 2, ch = (id & 3) * 8;
      *reinterpret_cast<uint4*>(&As[buf][row][ch]) = ra[i];
      *reinterpret_cast<uint4*>(&Bs[buf][row][ch]) = rb[i];
    }
  };
  fetch_tile(0);
  commit_tile(0);
  __syncthreads();
#endif

  v8f acc[2][4] = {};

  for (int kt = 0; kt < KT; ++kt) {
    const int cur = kt & 1;
#if ASYNC_LDS
    if (kt + 1 < KT) issue_tile((kt + 1) << 5, cur ^ 1);
#else
    if (kt + 1 < KT) fetch_tile((kt + 1) << 5);
#endif
    Frag a[2], b[4];
    const int kbA = lhalf * 8;              // A: K split {0..7,16..23}/{8..15,24..31}
#pragma unroll
    for (int tm = 0; tm < 2; ++tm) {
      int row = wm * 32 + tm * 16 + l16;
      a[tm].u[0] = *reinterpret_cast<const uint4*>(&As[cur][row][kbA]);
      a[tm].u[1] = *reinterpret_cast<const uint4*>(&As[cur][row][kbA + 16]);
    }
    const int kbB = lhalf * 8;              // B: K split {0..7,16..23}/{8..15,24..31}
#pragma unroll
    for (int tn = 0; tn < 4; ++tn) {
      int row = wn * 64 + tn * 16 + l16;
      b[tn].u[0] = *reinterpret_cast<const uint4*>(&Bs[cur][row][kbB]);
      b[tn].u[1] = *reinterpret_cast<const uint4*>(&Bs[cur][row][kbB + 16]);
    }
#pragma unroll
    for (int tm = 0; tm < 2; ++tm)
#pragma unroll
      for (int tn = 0; tn < 4; ++tn)
        acc[tm][tn] = __builtin_amdgcn_wmma_f32_16x16x32_bf16(
            false, a[tm].v, false, b[tn].v, (short)0, acc[tm][tn], false, false);

#if ASYNC_LDS
    async_wait0();
#else
    if (kt + 1 < KT) commit_tile(cur ^ 1);
#endif
    __syncthreads();
  }

#pragma unroll
  for (int tm = 0; tm < 2; ++tm)
#pragma unroll
    for (int tn = 0; tn < 4; ++tn)
#pragma unroll
      for (int r = 0; r < 8; ++r) {
        int row = m_blk + wm * 32 + tm * 16 + lhalf * 8 + r;
        int col = n_blk + wn * 64 + tn * 16 + l16;
        float v = acc[tm][tn][r];
        if (bias) v += bias[col];
        if (HAS_RES) v += res[(size_t)row * N + col];
        if (RELU) v = fmaxf(v, 0.0f);
        if (OUT_BF16)
          reinterpret_cast<bf16*>(outv)[(size_t)row * N + col] = f2bf(v);
        else
          reinterpret_cast<float*>(outv)[(size_t)row * N + col] = v;
      }
}

// ---------------------------------------------------------------------------
// Flash-style attention. Grid (S/128, B*H). qkv: [B,S,3*D] bf16; out [B,S,D].
// 128 queries/block, one head; each wave owns 16 query rows. K/V tiles are
// register-staged one iteration ahead so HBM loads overlap WMMA + softmax.
// ---------------------------------------------------------------------------
__global__ __launch_bounds__(256) void attention_wmma(
    const bf16* __restrict__ qkv, bf16* __restrict__ out) {
  constexpr int S = 2048, E = 3072, D = 1024;
  __shared__ bf16 Qs[128][72];
  __shared__ bf16 Ks[64][72];
  __shared__ bf16 Vt[64][72];   // transposed: [d][k]
  __shared__ bf16 Ps[128][72];  // per-wave 16-row strips
  const int t = threadIdx.x;
  const int wid = t >> 5, lane = t & 31;
  const int lhalf = lane >> 4, l16 = lane & 15;
  const int bh = blockIdx.y, b = bh >> 4, h = bh & 15;
  const int q0 = blockIdx.x * 128;
  const bf16* qbase = qkv + (size_t)b * S * E + h * 64;
  const bf16* kbase = qbase + D;
  const bf16* vbase = qbase + 2 * D;

#pragma unroll
  for (int i = 0; i < 4; ++i) {
    int id = t + 256 * i;
    int row = id >> 3, ch = (id & 7) * 8;
    *reinterpret_cast<uint4*>(&Qs[row][ch]) =
        *reinterpret_cast<const uint4*>(qbase + (size_t)(q0 + row) * E + ch);
  }

  uint4 rk[2], rv[2];
  auto fetch_kv = [&](int k0) {
#pragma unroll
    for (int i = 0; i < 2; ++i) {
      int id = t + 256 * i;
      int row = id >> 3, ch = (id & 7) * 8;
      rk[i] = *reinterpret_cast<const uint4*>(kbase + (size_t)(k0 + row) * E + ch);
      rv[i] = *reinterpret_cast<const uint4*>(vbase + (size_t)(k0 + row) * E + ch);
    }
  };
  fetch_kv(0);

  float m_run[8], l_run[8];
  v8f o[4] = {};
#pragma unroll
  for (int r = 0; r < 8; ++r) { m_run[r] = -1e30f; l_run[r] = 0.0f; }
  const float scale = 0.125f;  // 1/sqrt(64)

  for (int k0 = 0; k0 < S; k0 += 64) {
    __syncthreads();  // everyone done reading Ks/Vt from previous iteration
#pragma unroll
    for (int i = 0; i < 2; ++i) {
      int id = t + 256 * i;
      int row = id >> 3, ch = (id & 7) * 8;
      *reinterpret_cast<uint4*>(&Ks[row][ch]) = rk[i];
      union { uint4 u; bf16 e[8]; } tmp;
      tmp.u = rv[i];
#pragma unroll
      for (int j = 0; j < 8; ++j) Vt[ch + j][row] = tmp.e[j];
    }
    __syncthreads();
    if (k0 + 64 < S) fetch_kv(k0 + 64);   // overlap next HBM fetch with math

    // ---- S = Q K^T for this wave's 16 rows x 64 keys ----
    v8f sacc[4] = {};
#pragma unroll
    for (int s = 0; s < 2; ++s) {
      Frag a;
      const int kbA = s * 32 + lhalf * 8;
      const int arow = wid * 16 + l16;
      a.u[0] = *reinterpret_cast<const uint4*>(&Qs[arow][kbA]);
      a.u[1] = *reinterpret_cast<const uint4*>(&Qs[arow][kbA + 16]);
      const int kbB = s * 32 + lhalf * 8;
#pragma unroll
      for (int tn = 0; tn < 4; ++tn) {
        Frag bk;
        int row = tn * 16 + l16;
        bk.u[0] = *reinterpret_cast<const uint4*>(&Ks[row][kbB]);
        bk.u[1] = *reinterpret_cast<const uint4*>(&Ks[row][kbB + 16]);
        sacc[tn] = __builtin_amdgcn_wmma_f32_16x16x32_bf16(
            false, a.v, false, bk.v, (short)0, sacc[tn], false, false);
      }
    }

    // ---- online softmax (row = lhalf*8 + r, spread over 16 lanes) ----
    float mt[8], rs[8], alpha[8];
#pragma unroll
    for (int tn = 0; tn < 4; ++tn)
#pragma unroll
      for (int r = 0; r < 8; ++r) sacc[tn][r] = sacc[tn][r] * scale;
#pragma unroll
    for (int r = 0; r < 8; ++r) {
      float m = fmaxf(fmaxf(sacc[0][r], sacc[1][r]), fmaxf(sacc[2][r], sacc[3][r]));
#pragma unroll
      for (int off = 1; off < 16; off <<= 1) m = fmaxf(m, __shfl_xor(m, off, 32));
      mt[r] = m;
    }
#pragma unroll
    for (int r = 0; r < 8; ++r) {
      float mn = fmaxf(m_run[r], mt[r]);
      alpha[r] = __expf(m_run[r] - mn);
      m_run[r] = mn;
      rs[r] = 0.0f;
    }
#pragma unroll
    for (int tn = 0; tn < 4; ++tn)
#pragma unroll
      for (int r = 0; r < 8; ++r) {
        float p = __expf(sacc[tn][r] - m_run[r]);
        sacc[tn][r] = p;
        rs[r] += p;
      }
#pragma unroll
    for (int r = 0; r < 8; ++r) {
      float sm = rs[r];
#pragma unroll
      for (int off = 1; off < 16; off <<= 1) sm += __shfl_xor(sm, off, 32);
      l_run[r] = l_run[r] * alpha[r] + sm;
    }
#pragma unroll
    for (int tn = 0; tn < 4; ++tn)
#pragma unroll
      for (int r = 0; r < 8; ++r) o[tn][r] = o[tn][r] * alpha[r];

    // ---- stage P (wave-local strip), then O += P * V ----
#pragma unroll
    for (int tn = 0; tn < 4; ++tn)
#pragma unroll
      for (int r = 0; r < 8; ++r)
        Ps[wid * 16 + lhalf * 8 + r][tn * 16 + l16] = f2bf(sacc[tn][r]);

#pragma unroll
    for (int s = 0; s < 2; ++s) {
      Frag p;
      const int kbA = s * 32 + lhalf * 8;
      const int prow = wid * 16 + l16;
      p.u[0] = *reinterpret_cast<const uint4*>(&Ps[prow][kbA]);
      p.u[1] = *reinterpret_cast<const uint4*>(&Ps[prow][kbA + 16]);
      const int kbB = s * 32 + lhalf * 8;
#pragma unroll
      for (int tn = 0; tn < 4; ++tn) {
        Frag vb;
        int vrow = tn * 16 + l16;
        vb.u[0] = *reinterpret_cast<const uint4*>(&Vt[vrow][kbB]);
        vb.u[1] = *reinterpret_cast<const uint4*>(&Vt[vrow][kbB + 16]);
        o[tn] = __builtin_amdgcn_wmma_f32_16x16x32_bf16(
            false, p.v, false, vb.v, (short)0, o[tn], false, false);
      }
    }
  }

  bf16* obase = out + (size_t)b * S * D + h * 64;
#pragma unroll
  for (int tn = 0; tn < 4; ++tn)
#pragma unroll
    for (int r = 0; r < 8; ++r) {
      int row = q0 + wid * 16 + lhalf * 8 + r;
      int col = tn * 16 + l16;
      obase[(size_t)row * D + col] = f2bf(o[tn][r] / l_run[r]);
    }
}

// ---------------------------------------------------------------------------
// launcher
// ---------------------------------------------------------------------------
extern "C" void kernel_launch(void* const* d_in, const int* in_sizes, int n_in,
                              void* d_out, int out_size, void* d_ws, size_t ws_size,
                              hipStream_t stream) {
  constexpr int Bsz = 2, S = 2048, D = 1024, MS = Bsz * S;  // MS = 4096
  const float* x      = (const float*)d_in[0];
  const float* w_qkv  = (const float*)d_in[1];
  const float* w_proj = (const float*)d_in[2];
  const float* b_proj = (const float*)d_in[3];
  const float* w1     = (const float*)d_in[4];
  const float* b1     = (const float*)d_in[5];
  const float* w2     = (const float*)d_in[6];
  const float* b2     = (const float*)d_in[7];
  const float* ln1_g  = (const float*)d_in[8];
  const float* ln1_b  = (const float*)d_in[9];
  const float* ln2_g  = (const float*)d_in[10];
  const float* ln2_b  = (const float*)d_in[11];

  char* ws = (char*)d_ws;
  size_t off = 0;
  auto take = [&](size_t bytes) {
    char* p = ws + off;
    off += (bytes + 255) & ~(size_t)255;
    return p;
  };
  bf16*  wqkv_bf  = (bf16*)take((size_t)3 * D * D * 2);
  bf16*  wproj_bf = (bf16*)take((size_t)D * D * 2);
  bf16*  w1_bf    = (bf16*)take((size_t)4 * D * D * 2);
  bf16*  w2_bf    = (bf16*)take((size_t)4 * D * D * 2);
  bf16*  h_bf     = (bf16*)take((size_t)MS * D * 2);
  bf16*  qkv_bf   = (bf16*)take((size_t)MS * 3 * D * 2);
  bf16*  attn_bf  = (bf16*)take((size_t)MS * D * 2);
  float* x1       = (float*)take((size_t)MS * D * 4);
  bf16*  h2_bf    = (bf16*)take((size_t)MS * D * 2);
  bf16*  m1_bf    = (bf16*)take((size_t)MS * 4 * D * 2);

  auto cg = [](int n) { return dim3((unsigned)((n + 255) / 256)); };
  cast_f32_to_bf16<<<cg(3 * D * D), 256, 0, stream>>>(w_qkv, wqkv_bf, 3 * D * D);
  cast_f32_to_bf16<<<cg(D * D),     256, 0, stream>>>(w_proj, wproj_bf, D * D);
  cast_f32_to_bf16<<<cg(4 * D * D), 256, 0, stream>>>(w1, w1_bf, 4 * D * D);
  cast_f32_to_bf16<<<cg(4 * D * D), 256, 0, stream>>>(w2, w2_bf, 4 * D * D);

  // h = LN1(x)
  layernorm_to_bf16<<<MS, 256, 0, stream>>>(x, ln1_g, ln1_b, h_bf);
  // qkv = h @ w_qkv^T
  gemm_bf16_wmma<false, false, true><<<dim3(3 * D / 128, MS / 128), 256, 0, stream>>>(
      h_bf, wqkv_bf, nullptr, nullptr, qkv_bf, MS, 3 * D, D);
  // attn = softmax(QK^T/8) V
  attention_wmma<<<dim3(S / 128, Bsz * 16), 256, 0, stream>>>(qkv_bf, attn_bf);
  // x1 = x + attn @ w_proj^T + b_proj
  gemm_bf16_wmma<false, true, false><<<dim3(D / 128, MS / 128), 256, 0, stream>>>(
      attn_bf, wproj_bf, b_proj, x, x1, MS, D, D);
  // h2 = LN2(x1)
  layernorm_to_bf16<<<MS, 256, 0, stream>>>(x1, ln2_g, ln2_b, h2_bf);
  // m1 = relu(h2 @ w1^T + b1)
  gemm_bf16_wmma<true, false, true><<<dim3(4 * D / 128, MS / 128), 256, 0, stream>>>(
      h2_bf, w1_bf, b1, nullptr, m1_bf, MS, 4 * D, D);
  // out = x1 + m1 @ w2^T + b2
  gemm_bf16_wmma<false, true, false><<<dim3(D / 128, MS / 128), 256, 0, stream>>>(
      m1_bf, w2_bf, b2, x1, (float*)d_out, MS, D, 4 * D);

  (void)in_sizes; (void)n_in; (void)out_size; (void)ws_size;
}
